// TOODNet_41755672052335
// MI455X (gfx1250) — matrix operand order repaired and checked by reading, not compiled, not run on hardware
//
#include <hip/hip_runtime.h>

// ---------------------------------------------------------------------------
// CDNA5 WMMA types (wave32, v_wmma_f32_16x16x32_f16)
// ---------------------------------------------------------------------------
typedef __attribute__((ext_vector_type(16))) _Float16 v16h;
typedef __attribute__((ext_vector_type(8)))  _Float16 v8h;
typedef __attribute__((ext_vector_type(8)))  float    v8f;

#define NCLS 80

// ---------------------------------------------------------------------------
// Implicit-GEMM conv, Cin = 128 (fixed), activations f16 NHWC, weights f16
// pre-packed in WMMA A-fragment order. K ordering: k = (ky*KS+kx)*128 + c.
//   - one wave per block; each wave computes a 32(M) x 32(N) strip:
//     2x2 register tile of 16x16 D tiles -> 4 v_wmma per 32-K step with
//     only 4 x 32B loads (A fragments reused across N, B across M).
//   - A fragment: one contiguous 32B load per lane (packed layout keeps plain k
//     order; lane half h reads k 16h..16h+15 of each 32-K block, as does B).
//   - B fragment: lane needs channels c0+16*half .. +15 at one (n,y,x) ->
//     16 contiguous halves in NHWC -> one 32B load; padding taps use a
//     clamped address + vector select with zero.
// ---------------------------------------------------------------------------
template <int KS, bool OUTH>
__global__ __launch_bounds__(32) void conv_wmma(
    const _Float16* __restrict__ xh,   // input, f16 NHWC (Cin=128)
    const _Float16* __restrict__ wp,   // packed weights f16 [Cout][K]
    const float*    __restrict__ bias,
    float*          __restrict__ yf,   // f32 NHWC out (if !OUTH)
    _Float16*       __restrict__ yh,   // f16 NHWC out (if OUTH)
    int NB, int Cout, int H, int W, int relu)
{
    constexpr int CIN = 128;
    constexpr int K   = KS * KS * CIN;
    constexpr int PAD = (KS == 3) ? 1 : 0;
    const int HWp  = H * W;
    const int Ncol = NB * HWp;

    const int lane = threadIdx.x;
    const int half = lane >> 4;
    const int lid  = lane & 15;

    // Two N tiles (columns = flattened (n, y, x))
    int col[2]; bool cok[2]; int nn[2], oy[2], ox[2];
#pragma unroll
    for (int t = 0; t < 2; ++t) {
        col[t] = blockIdx.x * 32 + t * 16 + lid;
        cok[t] = (col[t] < Ncol);
        const int cc = cok[t] ? col[t] : (Ncol - 1);   // clamp for safe addr
        nn[t] = cc / HWp;
        const int r = cc % HWp;
        oy[t] = r / W;  ox[t] = r % W;
    }

    // Two M tiles (output channels), clamped for safe addressing.
    const int mrow0 = blockIdx.y * 32 + lid;
    const int mrow1 = mrow0 + 16;
    const int mc0 = (mrow0 < Cout) ? mrow0 : (Cout - 1);
    const int mc1 = (mrow1 < Cout) ? mrow1 : (Cout - 1);
    const _Float16* arow0 = wp + (size_t)mc0 * K + 16 * half;
    const _Float16* arow1 = wp + (size_t)mc1 * K + 16 * half;

    const v16h vzero = {};
    v8f acc[2][2] = {};   // [mtile][ntile]

#pragma unroll
    for (int tap = 0; tap < KS * KS; ++tap) {
        const int ky = tap / KS, kx = tap % KS;
        const _Float16* bptr[2]; bool bval[2];
#pragma unroll
        for (int t = 0; t < 2; ++t) {
            const int yy = oy[t] + ky - PAD;
            const int xx = ox[t] + kx - PAD;
            bval[t] = (yy >= 0) && (yy < H) && (xx >= 0) && (xx < W);
            const int yc = yy < 0 ? 0 : (yy > H - 1 ? H - 1 : yy);
            const int xc = xx < 0 ? 0 : (xx > W - 1 ? W - 1 : xx);
            bptr[t] = xh + ((size_t)(nn[t] * HWp + yc * W + xc)) * CIN + 16 * half;
        }
#pragma unroll
        for (int cb = 0; cb < CIN; cb += 32) {
            const int kb = tap * CIN + cb;
            const v16h a0 = *(const v16h*)(arow0 + kb);
            const v16h a1 = *(const v16h*)(arow1 + kb);
            const v16h b0 = bval[0] ? *(const v16h*)(bptr[0] + cb) : vzero;
            const v16h b1 = bval[1] ? *(const v16h*)(bptr[1] + cb) : vzero;
            acc[0][0] = __builtin_amdgcn_wmma_f32_16x16x32_f16(
                false, a0, false, b0, (short)0, acc[0][0], false, false);
            acc[0][1] = __builtin_amdgcn_wmma_f32_16x16x32_f16(
                false, a0, false, b1, (short)0, acc[0][1], false, false);
            acc[1][0] = __builtin_amdgcn_wmma_f32_16x16x32_f16(
                false, a1, false, b0, (short)0, acc[1][0], false, false);
            acc[1][1] = __builtin_amdgcn_wmma_f32_16x16x32_f16(
                false, a1, false, b1, (short)0, acc[1][1], false, false);
        }
    }

    // D tiles: VGPR r -> M = r + 8*half, N = lid; NHWC store, bias + ReLU.
    // 8 consecutive channels per lane -> vector stores when Cout % 8 == 0.
    const bool vec = ((Cout & 7) == 0);
#pragma unroll
    for (int mt = 0; mt < 2; ++mt) {
        const int co0 = blockIdx.y * 32 + mt * 16 + 8 * half;
#pragma unroll
        for (int t = 0; t < 2; ++t) {
            if (!cok[t]) continue;
            float vals[8];
#pragma unroll
            for (int r = 0; r < 8; ++r) {
                const int co = co0 + r;
                float vv = acc[mt][t][r] + ((co < Cout) ? bias[co] : 0.f);
                if (relu) vv = fmaxf(vv, 0.f);
                vals[r] = vv;
            }
            const size_t base = (size_t)col[t] * Cout + co0;
            if (vec && (co0 + 8 <= Cout)) {
                if (OUTH) {
                    v8h hv;
#pragma unroll
                    for (int r = 0; r < 8; ++r) hv[r] = (_Float16)vals[r];
                    *(v8h*)(yh + base) = hv;
                } else {
                    float4 v0 = make_float4(vals[0], vals[1], vals[2], vals[3]);
                    float4 v1 = make_float4(vals[4], vals[5], vals[6], vals[7]);
                    *(float4*)(yf + base)     = v0;
                    *(float4*)(yf + base + 4) = v1;
                }
            } else {
#pragma unroll
                for (int r = 0; r < 8; ++r) {
                    if (co0 + r < Cout) {
                        if (OUTH) yh[base + r] = (_Float16)vals[r];
                        else      yf[base + r] = vals[r];
                    }
                }
            }
        }
    }
}

// ---------------------------------------------------------------------------
// Weight repack: OIHW f32 -> f16 [cout][K] in WMMA A-fragment order.
// k-order: k = (ky*KS+kx)*128 + c, stored in plain k order (dst k = k) so
// each lane half reads 16 contiguous halves of each 32-wide K block.
// ---------------------------------------------------------------------------
__global__ void repack_w(const float* __restrict__ src, _Float16* __restrict__ dst,
                         int Cout, int KS)
{
    const int K = KS * KS * 128;
    const int d = blockIdx.x * blockDim.x + threadIdx.x;
    if (d >= Cout * K) return;
    const int cout = d / K;
    const int dk   = d % K;
    const int kb   = dk & ~31;
    const int p    = dk & 31;
    const int o    = p;
    const int k    = kb + o;
    const int tap  = k >> 7;
    const int c    = k & 127;
    const int ky   = tap / KS, kx = tap % KS;
    dst[d] = (_Float16)src[((size_t)(cout * 128 + c) * KS + ky) * KS + kx];
}

// feats: f32 NCHW -> f16 NHWC
__global__ void feat_to_nhwc(const float* __restrict__ src, _Float16* __restrict__ dst,
                             int HWp, int total)
{
    const int i = blockIdx.x * blockDim.x + threadIdx.x;
    if (i >= total) return;
    const int c    = i & 127;
    const int colp = i >> 7;           // n*HWp + pos
    const int n    = colp / HWp;
    const int pos  = colp % HWp;
    dst[i] = (_Float16)src[((size_t)(n * 128 + c)) * HWp + pos];
}

// GroupNorm stats over f32 NHWC conv output: block per (n, group);
// 4 channels/group are contiguous in NHWC -> one float4 per position.
__global__ void gn_stats(const float* __restrict__ x, float* __restrict__ stats,
                         int HWp)
{
    const int ng = blockIdx.x;         // n*32 + g
    const int n  = ng >> 5, g = ng & 31;
    float s = 0.f, s2 = 0.f;
    for (int pos = threadIdx.x; pos < HWp; pos += blockDim.x) {
        const float4 v = *(const float4*)(x + ((size_t)(n * HWp + pos) * 128 + g * 4));
        s  += v.x + v.y + v.z + v.w;
        s2 += v.x * v.x + v.y * v.y + v.z * v.z + v.w * v.w;
    }
    __shared__ float sh0[256], sh1[256];
    sh0[threadIdx.x] = s; sh1[threadIdx.x] = s2;
    __syncthreads();
    for (int off = 128; off > 0; off >>= 1) {
        if ((int)threadIdx.x < off) {
            sh0[threadIdx.x] += sh0[threadIdx.x + off];
            sh1[threadIdx.x] += sh1[threadIdx.x + off];
        }
        __syncthreads();
    }
    if (threadIdx.x == 0) {
        const float inv = 1.f / (float)(4 * HWp);
        const float mu  = sh0[0] * inv;
        const float var = sh1[0] * inv - mu * mu;
        stats[2 * ng]     = mu;
        stats[2 * ng + 1] = rsqrtf(var + 1e-5f);
    }
}

// GN apply + affine + ReLU, f32 NHWC -> f16 NHWC
__global__ void gn_apply(const float* __restrict__ x, const float* __restrict__ stats,
                         const float* __restrict__ gamma, const float* __restrict__ beta,
                         _Float16* __restrict__ yh, int HWp, int total)
{
    const int i = blockIdx.x * blockDim.x + threadIdx.x;
    if (i >= total) return;
    const int c    = i & 127;
    const int colp = i >> 7;
    const int n    = colp / HWp;
    const int g    = c >> 2;
    const float mu = stats[2 * (n * 32 + g)];
    const float rs = stats[2 * (n * 32 + g) + 1];
    const float v  = (x[i] - mu) * rs * gamma[c] + beta[c];
    yh[i] = (_Float16)fmaxf(v, 0.f);
}

// sqrt(sigmoid(cl)*sigmoid(cp)) -> cls output rows (NHWC in, row-major out)
__global__ void cls_combine(const float* __restrict__ cl, const float* __restrict__ cp2,
                            float* __restrict__ out_cls, int HWp, int total,
                            int rowBase, int Rtot)
{
    const int i = blockIdx.x * blockDim.x + threadIdx.x;
    if (i >= total) return;
    const int cc   = i % NCLS;
    const int colp = i / NCLS;
    const int n    = colp / HWp;
    const int pos  = colp % HWp;
    const float a  = cl[(size_t)colp * NCLS + cc];
    const float b  = cp2[colp];
    const float sa = 1.f / (1.f + __expf(-a));
    const float sb = 1.f / (1.f + __expf(-b));
    out_cls[((size_t)n * Rtot + rowBase + pos) * NCLS + cc] = sqrtf(sa * sb);
}

__device__ __forceinline__ float bilin_tap4(const float* brn, float yy, float xx,
                                            int H, int W, int c)
{
    if (yy < 0.f || yy > (float)(H - 1) || xx < 0.f || xx > (float)(W - 1)) return 0.f;
    const int yc = (int)fminf(fmaxf(yy, 0.f), (float)(H - 1));
    const int xc = (int)fminf(fmaxf(xx, 0.f), (float)(W - 1));
    return brn[(size_t)(yc * W + xc) * 4 + c];
}

// dcn_ones_1x1 over NHWC br (4ch) / off (8ch): bilinear sample + channel sum,
// broadcast x4 into box output rows.
__global__ void dcn_box(const float* __restrict__ br, const float* __restrict__ off,
                        float* __restrict__ out_box, int NB, int H, int W,
                        int rowBase, int Rtot)
{
    const int HWp = H * W;
    const int idx = blockIdx.x * blockDim.x + threadIdx.x;
    if (idx >= NB * HWp) return;
    const int n   = idx / HWp;
    const int pos = idx % HWp;
    const int oy  = pos / W;
    const int ox  = pos % W;
    const float* brn = br + (size_t)n * HWp * 4;

    float s = 0.f;
#pragma unroll
    for (int c = 0; c < 4; ++c) {
        const float dy = off[(size_t)idx * 8 + 2 * c];
        const float dx = off[(size_t)idx * 8 + 2 * c + 1];
        const float py = (float)oy + dy;
        const float px = (float)ox + dx;
        const float fy = floorf(py), fx = floorf(px);
        const float wy = py - fy,    wx = px - fx;
        s += bilin_tap4(brn, fy,       fx,       H, W, c) * (1.f - wy) * (1.f - wx)
           + bilin_tap4(brn, fy,       fx + 1.f, H, W, c) * (1.f - wy) * wx
           + bilin_tap4(brn, fy + 1.f, fx,       H, W, c) * wy * (1.f - wx)
           + bilin_tap4(brn, fy + 1.f, fx + 1.f, H, W, c) * wy * wx;
    }
    const size_t row = ((size_t)n * Rtot + rowBase + pos) * 4;
    out_box[row + 0] = s; out_box[row + 1] = s;
    out_box[row + 2] = s; out_box[row + 3] = s;
}

__global__ void ctr_write(const float* __restrict__ ctr, float* __restrict__ out_ctr,
                          int NB, int HWp, int rowBase, int Rtot)
{
    const int idx = blockIdx.x * blockDim.x + threadIdx.x;
    if (idx >= NB * HWp) return;
    const int n   = idx / HWp;
    const int pos = idx % HWp;
    out_ctr[(size_t)n * Rtot + rowBase + pos] = ctr[idx];
}

// ---------------------------------------------------------------------------
// Host orchestration (graph-capture safe)
// ---------------------------------------------------------------------------
extern "C" void kernel_launch(void* const* d_in, const int* in_sizes, int n_in,
                              void* d_out, int out_size, void* d_ws, size_t ws_size,
                              hipStream_t stream)
{
    (void)in_sizes; (void)n_in; (void)out_size; (void)ws_size;

    const int Hs[5] = {64, 32, 16, 8, 4};
    const int Ws[5] = {100, 50, 25, 13, 7};
    const int NB = 2, C = 128;

    const float* feats[5];
    for (int i = 0; i < 5; ++i) feats[i] = (const float*)d_in[i];
    const float* tower_w    = (const float*)d_in[5];
    const float* tower_b    = (const float*)d_in[6];
    const float* tower_g    = (const float*)d_in[7];
    const float* tower_beta = (const float*)d_in[8];
    const float* cls_w  = (const float*)d_in[9];
    const float* cls_b  = (const float*)d_in[10];
    const float* cp1_w  = (const float*)d_in[11];
    const float* cp1_b  = (const float*)d_in[12];
    const float* cp2_w  = (const float*)d_in[13];
    const float* cp2_b  = (const float*)d_in[14];
    const float* box_w  = (const float*)d_in[15];
    const float* box_b  = (const float*)d_in[16];
    const float* off1_w = (const float*)d_in[17];
    const float* off1_b = (const float*)d_in[18];
    const float* off2_w = (const float*)d_in[19];
    const float* off2_b = (const float*)d_in[20];
    const float* ctr_w  = (const float*)d_in[21];
    const float* ctr_b  = (const float*)d_in[22];

    // ---- workspace carve-up, 256B-aligned regions ----
    char* wsb = (char*)d_ws;
    size_t off = 0;
    auto carve = [&](size_t bytes) -> char* {
        char* p = wsb + off;
        off = (off + bytes + 255) & ~(size_t)255;
        return p;
    };
    const size_t MAXHW  = (size_t)NB * 64 * 100;      // 12,800 positions
    const size_t MAXMAP = MAXHW * C;                  // 1,638,400 elements
    const int K3 = 9 * 128, K1 = 128;

    _Float16* wpTower = (_Float16*)carve(6 * (size_t)C * K3 * 2);
    _Float16* wpCls   = (_Float16*)carve((size_t)NCLS * K3 * 2);
    _Float16* wpCp1   = (_Float16*)carve((size_t)C * K1 * 2);
    _Float16* wpCp2   = (_Float16*)carve((size_t)1 * K3 * 2);
    _Float16* wpBox   = (_Float16*)carve((size_t)4 * K3 * 2);
    _Float16* wpOff1  = (_Float16*)carve((size_t)C * K1 * 2);
    _Float16* wpOff2  = (_Float16*)carve((size_t)8 * K3 * 2);
    _Float16* wpCtr   = (_Float16*)carve((size_t)1 * K3 * 2);

    _Float16* hF    = (_Float16*)carve(MAXMAP * 2);   // current tower feature
    _Float16* hCP1  = (_Float16*)carve(MAXMAP * 2);
    _Float16* hOFF1 = (_Float16*)carve(MAXMAP * 2);
    float* fCONV = (float*)carve(MAXMAP * 4);         // tower conv pre-GN
    float* fCL   = (float*)carve((size_t)NCLS * MAXHW * 4);
    float* fBOX  = (float*)carve(4 * MAXHW * 4);
    float* fOFF2 = (float*)carve(8 * MAXHW * 4);
    float* fCP2  = (float*)carve(MAXHW * 4);
    float* fCTR  = (float*)carve(MAXHW * 4);
    float* stats = (float*)carve(128 * 4);

    // ---- one-time weight repack (f32 OIHW -> f16 WMMA-packed) ----
    auto rp = [&](const float* src, _Float16* dst, int Cout, int KS) {
        const int tot = Cout * KS * KS * 128;
        repack_w<<<(tot + 255) / 256, 256, 0, stream>>>(src, dst, Cout, KS);
    };
    for (int it = 0; it < 6; ++it)
        rp(tower_w + (size_t)it * C * C * 9, wpTower + (size_t)it * C * K3, C, 3);
    rp(cls_w, wpCls, NCLS, 3);
    rp(cp1_w, wpCp1, C, 1);
    rp(cp2_w, wpCp2, 1, 3);
    rp(box_w, wpBox, 4, 3);
    rp(off1_w, wpOff1, C, 1);
    rp(off2_w, wpOff2, 8, 3);
    rp(ctr_w, wpCtr, 1, 3);

    // ---- output segment bookkeeping ----
    int cumHW[5], cum = 0;
    for (int li = 0; li < 5; ++li) { cumHW[li] = cum; cum += Hs[li] * Ws[li]; }
    const int Rtot = 6 * cum;                         // 51192 rows / batch
    float* out_cls = (float*)d_out;
    float* out_box = out_cls + (size_t)NB * Rtot * NCLS;
    float* out_ctr = out_box + (size_t)NB * Rtot * 4;

    const dim3 wave(32);

    for (int li = 0; li < 5; ++li) {
        const int H = Hs[li], W = Ws[li], HWp = H * W;
        const int Ncol = NB * HWp;
        const int gN = (Ncol + 31) / 32;
        const int totE = NB * HWp * C;

        feat_to_nhwc<<<(totE + 255) / 256, 256, 0, stream>>>(feats[li], hF, HWp, totE);

        for (int it = 0; it < 6; ++it) {
            const int rowBase = 6 * cumHW[li] + it * HWp;

            // ---- tower: conv3x3(128->128) + GN + ReLU (f -> hF) ----
            conv_wmma<3, false><<<dim3(gN, 4), wave, 0, stream>>>(
                hF, wpTower + (size_t)it * C * K3, tower_b + it * C,
                fCONV, nullptr, NB, C, H, W, 0);
            gn_stats<<<NB * 32, 256, 0, stream>>>(fCONV, stats, HWp);
            gn_apply<<<(totE + 255) / 256, 256, 0, stream>>>(
                fCONV, stats, tower_g + it * C, tower_beta + it * C, hF, HWp, totE);

            // ---- cls branch ----
            conv_wmma<1, true><<<dim3(gN, 4), wave, 0, stream>>>(
                hF, wpCp1, cp1_b, nullptr, hCP1, NB, C, H, W, 1);
            conv_wmma<3, false><<<dim3(gN, 1), wave, 0, stream>>>(
                hCP1, wpCp2, cp2_b, fCP2, nullptr, NB, 1, H, W, 0);
            conv_wmma<3, false><<<dim3(gN, 3), wave, 0, stream>>>(
                hF, wpCls, cls_b, fCL, nullptr, NB, NCLS, H, W, 0);
            {
                const int tot = NB * HWp * NCLS;
                cls_combine<<<(tot + 255) / 256, 256, 0, stream>>>(
                    fCL, fCP2, out_cls, HWp, tot, rowBase, Rtot);
            }

            // ---- box branch ----
            conv_wmma<3, false><<<dim3(gN, 1), wave, 0, stream>>>(
                hF, wpBox, box_b, fBOX, nullptr, NB, 4, H, W, 1);
            conv_wmma<1, true><<<dim3(gN, 4), wave, 0, stream>>>(
                hF, wpOff1, off1_b, nullptr, hOFF1, NB, C, H, W, 1);
            conv_wmma<3, false><<<dim3(gN, 1), wave, 0, stream>>>(
                hOFF1, wpOff2, off2_b, fOFF2, nullptr, NB, 8, H, W, 0);
            dcn_box<<<(NB * HWp + 255) / 256, 256, 0, stream>>>(
                fBOX, fOFF2, out_box, NB, H, W, rowBase, Rtot);

            // ---- centerness ----
            conv_wmma<3, false><<<dim3(gN, 1), wave, 0, stream>>>(
                hF, wpCtr, ctr_b, fCTR, nullptr, NB, 1, H, W, 0);
            ctr_write<<<(NB * HWp + 255) / 256, 256, 0, stream>>>(
                fCTR, out_ctr, NB, HWp, rowBase, Rtot);
        }
    }
}
